// CConvAggregationBlock_21698174779789
// MI455X (gfx1250) — matrix operand order repaired and checked by reading, not compiled, not run on hardware
//
#include <hip/hip_runtime.h>
#include <hip/hip_bf16.h>

typedef __attribute__((ext_vector_type(16))) _Float16 v16h;
typedef __attribute__((ext_vector_type(8)))  float    v8f;

#define CC_N_OUT   50000
#define CC_IN_CH   32
#define CC_OUT_CH  64
#define CC_K3      64
#define CC_KDIM    2048          // K3 * IN_CH
#define TILE_M     16
#define ROWP       2050          // padded LDS row stride in floats (even, breaks bank conflicts)

// ---------------------------------------------------------------------------
// Phase 0: pack W [2048,64] fp32 -> f16 B-matrix fragments for
// v_wmma_f32_16x16x32_f16.  B packing used here (32x16, f16, wave32; contiguous k order, matches A):
//   lane L: column n = L&15, half hi = L>>4 ; VGPR v holds K pair
//   k = hi*16 + 2v  ->  B[k][n], B[k+1][n]
// Packed as Wh[((kk*4 + nt)*32 + lane)*16 + 2v {+1}] so the main kernel's B
// operand is one contiguous v16h (32B) load per WMMA, L2-resident (256 KB).
// ---------------------------------------------------------------------------
__global__ void cconv_pack_w_kernel(const float* __restrict__ W,
                                    _Float16* __restrict__ Wh) {
  int t = blockIdx.x * blockDim.x + threadIdx.x;   // 65536 K-pair slots
  if (t >= 64 * 4 * 32 * 8) return;
  int v    = t & 7;
  int lane = (t >> 3) & 31;
  int nt   = (t >> 8) & 3;
  int kk   = t >> 10;                              // 0..63
  int n    = nt * 16 + (lane & 15);
  int hi   = lane >> 4;
  int k    = kk * 32 + hi * 16 + 2 * v;
  size_t base = ((size_t)((kk * 4 + nt) * 32 + lane)) * 16 + 2 * v;
  Wh[base]     = (_Float16)W[(size_t)k       * CC_OUT_CH + n];
  Wh[base + 1] = (_Float16)W[(size_t)(k + 1) * CC_OUT_CH + n];
}

// ---------------------------------------------------------------------------
// Fused kernel: one block = 16 output points, 128 threads (4 waves).
// Stage A: scatter edges into LDS A tile [16][2048] fp32 via ds_add_f32.
// Stage B: WMMA GEMM  A[16,2048] x W[2048,64] -> y[16,64], normalize+bias+relu.
// ---------------------------------------------------------------------------
__global__ __launch_bounds__(128)
void cconv_fused_kernel(const float* __restrict__ feats,
                        const float* __restrict__ inp_points,
                        const float* __restrict__ out_points,
                        const float* __restrict__ out_extents,
                        const float* __restrict__ scale_compat,
                        const int*   __restrict__ nidx,
                        const int*   __restrict__ row_splits,
                        const float* __restrict__ ndist,
                        const _Float16* __restrict__ Wh,
                        const float* __restrict__ bias,
                        float* __restrict__ out) {
  __shared__ float Af[TILE_M * ROWP];     // 131200 B
  __shared__ float denom_s[TILE_M];
  __shared__ float invd_s[TILE_M];
  __shared__ int   rsp[TILE_M + 1];

  const int tid  = threadIdx.x;
  const int lane = tid & 31;
  const int wv   = tid >> 5;              // wave id 0..3
  const int j0   = blockIdx.x * TILE_M;

  for (int i = tid; i < TILE_M * ROWP; i += 128) Af[i] = 0.0f;
  if (tid < TILE_M)  denom_s[tid] = 0.0f;
  if (tid <= TILE_M) rsp[tid] = row_splits[j0 + tid];
  __syncthreads();

  const float s2 = 2.0f / out_extents[0];
  // readfirstlane -> edge index is provably wave-uniform, so per-edge scalar
  // data (nidx/ndist/scale_compat/points) lowers to SMEM broadcast loads.
  const int e0 = __builtin_amdgcn_readfirstlane(rsp[0]);
  const int e1 = __builtin_amdgcn_readfirstlane(rsp[TILE_M]);

  // -------- Stage A: edge scatter (one wave per edge; lane = channel) --------
  for (int e = e0 + wv; e < e1; e += 4) {
    // local destination row = last m with rsp[m] <= e  (searchsorted 'right'-1)
    int m = 0;
#pragma unroll
    for (int t = 1; t < TILE_M; ++t) m += (e >= rsp[t]) ? 1 : 0;

    const int nbr = nidx[e];
    const float px = inp_points[3 * nbr + 0];
    const float py = inp_points[3 * nbr + 1];
    const float pz = inp_points[3 * nbr + 2];
    const int jo = j0 + m;
    const float rx = (px - out_points[3 * jo + 0]) * s2;
    const float ry = (py - out_points[3 * jo + 1]) * s2;
    const float rz = (pz - out_points[3 * jo + 2]) * s2;

    const float r   = sqrtf(rx * rx + ry * ry + rz * rz);
    const float inf = fmaxf(fabsf(rx), fmaxf(fabsf(ry), fabsf(rz)));
    const float scl = (inf > 1e-8f) ? (r / fmaxf(inf, 1e-8f)) : 0.0f;

    const float ux = fminf(fmaxf(rx * scl, -1.0f), 1.0f);
    const float uy = fminf(fmaxf(ry * scl, -1.0f), 1.0f);
    const float uz = fminf(fmaxf(rz * scl, -1.0f), 1.0f);

    const float tx = (ux + 1.0f) * 1.5f;
    const float ty = (uy + 1.0f) * 1.5f;
    const float tz = (uz + 1.0f) * 1.5f;
    const float fx = fminf(fmaxf(floorf(tx), 0.0f), 2.0f);
    const float fy = fminf(fmaxf(floorf(ty), 0.0f), 2.0f);
    const float fz = fminf(fmaxf(floorf(tz), 0.0f), 2.0f);
    const float gx = tx - fx, gy = ty - fy, gz = tz - fz;
    const int ix = (int)fx, iy = (int)fy, iz = (int)fz;

    const float d  = ndist[e];
    const float q  = 1.0f - d * d;
    const float w6 = fminf(fmaxf(q * q * q, 0.0f), 1.0f);
    const float imp = scale_compat[e] * w6;

    const float fwc = feats[(size_t)nbr * CC_IN_CH + lane] * imp;
    const float wx[2] = {1.0f - gx, gx};
    const float wy[2] = {1.0f - gy, gy};
    const float wz[2] = {1.0f - gz, gz};

#pragma unroll
    for (int dx = 0; dx < 2; ++dx)
#pragma unroll
      for (int dy = 0; dy < 2; ++dy)
#pragma unroll
        for (int dz = 0; dz < 2; ++dz) {
          const int kbin = ((ix + dx) * 4 + iy + dy) * 4 + iz + dz;
          const float w = wx[dx] * wy[dy] * wz[dz];
          atomicAdd(&Af[m * ROWP + kbin * CC_IN_CH + lane], fwc * w);
        }
    if (lane == 0) atomicAdd(&denom_s[m], imp);
  }
  __syncthreads();

  // Invert the 16 row denominators once (instead of 8 IEEE divides per thread
  // in the epilogue).
  if (tid < TILE_M) {
    const float dn = denom_s[tid];
    invd_s[tid] = (dn != 0.0f) ? (1.0f / dn) : 1.0f;
  }
  __syncthreads();

  // -------- Stage B: WMMA GEMM, wave wv owns N-tile columns [wv*16, wv*16+16) --------
  const int nt   = wv;
  const int mrow = lane & 15;
  const int hi   = lane >> 4;
  const v16h* Bp = ((const v16h*)Wh) + (nt * 32 + lane);
  const float* Arow = &Af[mrow * ROWP];

  v8f c = {};
  for (int kk = 0; kk < 64; ++kk) {
    v16h a;
#pragma unroll
    for (int v = 0; v < 8; ++v) {
      // A operand in this code's contiguous k order: VGPR v pair K = hi*16 + 2*v (matches packed B)
      const int k = kk * 32 + hi * 16 + (v << 1);
      const float2 f2 = *(const float2*)(Arow + k);
      a[2 * v]     = (_Float16)f2.x;
      a[2 * v + 1] = (_Float16)f2.y;
    }
    const v16h b = Bp[(size_t)kk * 128];
    c = __builtin_amdgcn_wmma_f32_16x16x32_f16(
        /*neg_a=*/false, a, /*neg_b=*/false, b,
        /*c_mod=*/(short)0, c, /*reuse_a=*/false, /*reuse_b=*/false);
  }

  // C/D layout: VGPR r -> row m = r + 8*hi, col n = lane&15 (within N-tile)
#pragma unroll
  for (int r = 0; r < 8; ++r) {
    const int m = r + 8 * hi;
    const int n = nt * 16 + mrow;
    const float val = c[r] * invd_s[m] + bias[n];
    out[(size_t)(j0 + m) * CC_OUT_CH + n] = fmaxf(val, 0.0f);
  }
}

// ---------------------------------------------------------------------------
extern "C" void kernel_launch(void* const* d_in, const int* in_sizes, int n_in,
                              void* d_out, int out_size, void* d_ws, size_t ws_size,
                              hipStream_t stream) {
  const float* feats        = (const float*)d_in[0];
  const float* inp_points   = (const float*)d_in[1];
  const float* out_points   = (const float*)d_in[2];
  const float* out_extents  = (const float*)d_in[3];
  const float* scale_compat = (const float*)d_in[4];
  const int*   nidx         = (const int*)d_in[5];
  const int*   row_splits   = (const int*)d_in[6];
  const float* ndist        = (const float*)d_in[7];
  const float* W            = (const float*)d_in[8];
  const float* bias         = (const float*)d_in[9];
  float*       out          = (float*)d_out;
  _Float16*    Wh           = (_Float16*)d_ws;   // 256 KB of fragment-packed f16 W

  // Phase 0: pack W into WMMA B-fragment layout (f16).
  cconv_pack_w_kernel<<<256, 256, 0, stream>>>(W, Wh);

  // Fused scatter + WMMA GEMM: 50000 / 16 = 3125 blocks exactly.
  const int nblocks = CC_N_OUT / TILE_M;
  cconv_fused_kernel<<<nblocks, 128, 0, stream>>>(
      feats, inp_points, out_points, out_extents, scale_compat,
      nidx, row_splits, ndist, Wh, bias, out);
}
